// Qwen3MegaBlocksAdapter_58858231824406
// MI455X (gfx1250) — matrix operand order repaired and checked by reading, not compiled, not run on hardware
//
#include <hip/hip_runtime.h>
#include <hip/hip_bf16.h>

// ---------------- problem constants (from reference) ----------------
constexpr int E    = 32;    // experts
constexpr int TOPK = 8;     // top-k
constexpr int H    = 2048;  // hidden
constexpr int F    = 768;   // intermediate
constexpr int T    = 1024;  // tokens (B*S)
constexpr int KC   = 512;   // K-chunk staged in LDS for x

// tile bookkeeping: WMMA K-tile = 32, N-tile = 16
constexpr int NT1 = F / 16;   // 48  N-tiles for W1/V1
constexpr int KT1 = H / 32;   // 64  K-tiles for W1/V1
constexpr int NT2 = H / 16;   // 128 N-tiles for W2
constexpr int KT2 = F / 32;   // 24  K-tiles for W2
constexpr size_t FRAGS_PER_MAT = (size_t)E * NT1 * KT1;  // == E*NT2*KT2 == 98304

typedef __attribute__((ext_vector_type(8)))  float  v8f;
typedef __attribute__((ext_vector_type(16))) __bf16 v16bf;

union FragU { unsigned short s[16]; uint4 q[2]; v16bf v; };

// round-to-nearest-even f32 -> bf16 (bit pattern as ushort)
static __device__ __forceinline__ unsigned short f2bf(float f) {
    union { float f; unsigned int u; } x; x.f = f;
    unsigned int r = x.u + 0x7FFFu + ((x.u >> 16) & 1u);
    return (unsigned short)(r >> 16);
}

static __device__ __forceinline__ v8f wmma_bf16(v16bf a, v16bf b, v8f c) {
    return __builtin_amdgcn_wmma_f32_16x16x32_bf16(
        false, a, false, b, (short)0, c, false, false);
}

// A-fragment (16x32, M x K) from LDS bf16, contiguous K order (matches B):
//   lanes 0-15 : row M=lane,   K = k0 + 0..15
//   lanes16-31 : row M=lane-16,K = k0 + 16..31
static __device__ __forceinline__ v16bf load_a_lds(const unsigned short* base,
                                                   int ldk, int k0) {
    int lane = threadIdx.x & 31;
    int m    = lane & 15;
    int kb   = (lane >> 4) << 4;          // 0 or 16
    const unsigned short* p = base + m * ldk + k0 + kb;
    FragU u;
    u.q[0] = *(const uint4*)(p);          // K = kb+0..7
    u.q[1] = *(const uint4*)(p + 8);      // K = kb+8..15
    return u.v;
}

// B-fragment (32x16, K x N) gathered from f32 global weights (row-major, ld=ldn)
// into WMMA lane order:
//   lanes 0-15 : col N=lane,    K = k0 + 0..15
//   lanes16-31 : col N=lane-16, K = k0 + 16..31
static __device__ __forceinline__ void fetch_b_f32(const float* __restrict__ W,
                                                   int ldn, int k0, int n0,
                                                   FragU& u) {
    int lane = threadIdx.x & 31;
    int n    = lane & 15;
    int kb   = (lane >> 4) << 4;
    const float* p = W + (size_t)(k0 + kb) * ldn + n0 + n;
#pragma unroll
    for (int i = 0; i < 16; ++i)
        u.s[i] = f2bf(p[(size_t)i * ldn]);
}

static __device__ __forceinline__ v16bf load_b_global(const float* __restrict__ W,
                                                      int ldn, int k0, int n0) {
    FragU u;
    fetch_b_f32(W, ldn, k0, n0, u);
    return u.v;
}

// B-fragment from pre-shuffled bf16 buffer: lane l owns 32 contiguous bytes.
static __device__ __forceinline__ v16bf load_b_tiled(const unsigned short* __restrict__ frag) {
    int lane = threadIdx.x & 31;
    const uint4* p = (const uint4*)(frag + lane * 16);
    FragU u;
    u.q[0] = p[0];
    u.q[1] = p[1];
    return u.v;
}

// ---------------- kernel: zero output (we scatter-add into it) ----------
__global__ void k_zero(float* __restrict__ out, int n) {
    int i = blockIdx.x * blockDim.x + threadIdx.x;
    if (i < n) out[i] = 0.f;
}

// ---------------- kernel: x -> bf16 ----------------
__global__ void k_cvt_x(const float* __restrict__ x, unsigned short* __restrict__ xb, int n) {
    int i4 = (blockIdx.x * blockDim.x + threadIdx.x) * 4;
    if (i4 >= n) return;
    float4 v = *(const float4*)(x + i4);
    uint2 o;
    o.x = (unsigned)f2bf(v.x) | ((unsigned)f2bf(v.y) << 16);
    o.y = (unsigned)f2bf(v.z) | ((unsigned)f2bf(v.w) << 16);
    *(uint2*)(xb + i4) = o;
}

// ---------------- kernel: weights f32 -> bf16, shuffled to fragment order ----
// One wave per 32x16 WMMA tile. Fragment layout on disk:
//   W1/V1: frag[(e*NT1 + nt)*KT1 + kt], W2: frag[(e*NT2 + nt)*KT2 + kt],
//   each fragment = 512 bf16 (1 KB), lane l's 16 values contiguous at l*32 B.
__global__ void k_convert(const float* __restrict__ w1, const float* __restrict__ v1,
                          const float* __restrict__ w2,
                          unsigned short* __restrict__ w1b, unsigned short* __restrict__ v1b,
                          unsigned short* __restrict__ w2b) {
    size_t wid = (size_t)blockIdx.x * 8 + (threadIdx.x >> 5);
    if (wid >= 3 * FRAGS_PER_MAT) return;
    int lane = threadIdx.x & 31;

    const float* src; unsigned short* dst; int ldn, k0, n0;
    int m   = (int)(wid / FRAGS_PER_MAT);
    size_t rem = wid % FRAGS_PER_MAT;
    if (m < 2) {
        int e  = (int)(rem / (NT1 * KT1));
        int r2 = (int)(rem % (NT1 * KT1));
        int nt = r2 / KT1, kt = r2 % KT1;
        src = (m == 0 ? w1 : v1) + (size_t)e * H * F;
        dst = (m == 0 ? w1b : v1b) + rem * 512;
        ldn = F; k0 = kt * 32; n0 = nt * 16;
    } else {
        int e  = (int)(rem / (NT2 * KT2));
        int r2 = (int)(rem % (NT2 * KT2));
        int nt = r2 / KT2, kt = r2 % KT2;
        src = w2 + (size_t)e * F * H;
        dst = w2b + rem * 512;
        ldn = H; k0 = kt * 32; n0 = nt * 16;
    }
    FragU u;
    fetch_b_f32(src, ldn, k0, n0, u);
    uint4* d = (uint4*)(dst + lane * 16);
    d[0] = u.q[0];
    d[1] = u.q[1];
}

// ---------------- kernel: router ----------------
__global__ void k_router(const float* __restrict__ x, const float* __restrict__ wg,
                         int* __restrict__ topk_idx, float* __restrict__ topk_w) {
    int wave = threadIdx.x >> 5;
    int lane = threadIdx.x & 31;
    int t    = blockIdx.x * 8 + wave;
    const float* xr = x  + (size_t)t    * H;
    const float* wr = wg + (size_t)lane * H;
    float acc = 0.f;
    for (int h = 0; h < H; h += 4) {
        float4 xv = *(const float4*)(xr + h);
        float4 wv = *(const float4*)(wr + h);
        acc += xv.x * wv.x + xv.y * wv.y + xv.z * wv.z + xv.w * wv.w;
    }
    float m = acc;
    for (int off = 16; off > 0; off >>= 1) m = fmaxf(m, __shfl_xor(m, off, 32));
    float p = __expf(acc - m);
    float s = p;
    for (int off = 16; off > 0; off >>= 1) s += __shfl_xor(s, off, 32);
    float prob = p / s;

    float wk[TOPK]; int ik[TOPK];
    float cur = prob;
#pragma unroll
    for (int k = 0; k < TOPK; ++k) {
        float v = cur; int ix = lane;
        for (int off = 16; off > 0; off >>= 1) {
            float ov = __shfl_xor(v, off, 32);
            int   oi = __shfl_xor(ix, off, 32);
            if (ov > v || (ov == v && oi < ix)) { v = ov; ix = oi; }
        }
        wk[k] = v; ik[k] = ix;
        if (lane == ix) cur = -1e30f;
    }
    float l1 = 0.f;
#pragma unroll
    for (int k = 0; k < TOPK; ++k) l1 += wk[k];
    if (lane == 0) {
#pragma unroll
        for (int k = 0; k < TOPK; ++k) {
            topk_idx[t * TOPK + k] = ik[k];
            topk_w  [t * TOPK + k] = wk[k] / l1;
        }
    }
}

// ---------------- kernel: deterministic per-expert token lists ----------
__global__ void k_build(const int* __restrict__ topk_idx, const float* __restrict__ topk_w,
                        int* __restrict__ counts, int* __restrict__ ltok,
                        float* __restrict__ lw) {
    int e = threadIdx.x;
    if (e >= E) return;
    int cnt = 0;
    for (int p = 0; p < T * TOPK; ++p) {
        if (topk_idx[p] == e) {
            ltok[e * T + cnt] = p / TOPK;
            lw  [e * T + cnt] = topk_w[p];
            ++cnt;
        }
    }
    counts[e] = cnt;
}

// ---------------- fast path: fused expert GLU-MLP on pre-shuffled bf16 ------
// grid.x = E*64 ; expert e = bx>>6, token-tile tb = bx&63 (16 tokens).
// 256 threads = 8 waves. Inner loop: 2x ds_load_b128 (A) + 4x global_load_b128
// (pre-tiled B) + 2x v_wmma per K-step.
__launch_bounds__(256)
__global__ void k_expert_pre(const unsigned short* __restrict__ xb,
                             const unsigned short* __restrict__ w1b,
                             const unsigned short* __restrict__ v1b,
                             const unsigned short* __restrict__ w2b,
                             const int* __restrict__ counts, const int* __restrict__ ltok,
                             const float* __restrict__ lw, float* __restrict__ out) {
    __shared__ __align__(16) unsigned short As[16 * KC];   // 16 KB
    __shared__ __align__(16) unsigned short Hs[16 * F];    // 24 KB
    __shared__ int   stok[16];
    __shared__ float sw[16];

    int e  = blockIdx.x >> 6;
    int tb = blockIdx.x & 63;
    int cnt   = counts[e];
    int tiles = (cnt + 15) >> 4;
    if (tb >= tiles) return;

    int tid = threadIdx.x;
    if (tid < 16) {
        int p = tb * 16 + tid;
        if (p < cnt) { stok[tid] = ltok[e * T + p]; sw[tid] = lw[e * T + p]; }
        else         { stok[tid] = 0;               sw[tid] = 0.f; }
    }
    __syncthreads();

    int wave = tid >> 5;
    int lane = tid & 31;
    int mrow = (lane >> 4) << 3;
    int ncol = lane & 15;

    v8f accg[6], accu[6];
    v8f vzero = {};
#pragma unroll
    for (int i = 0; i < 6; ++i) { accg[i] = vzero; accu[i] = vzero; }

    for (int kc = 0; kc < H; kc += KC) {
        // stage A (already bf16): pure 16B copies
        for (int i = tid; i < (16 * KC) / 8; i += 256) {
            int r  = i >> 6;              // KC/8 == 64
            int k8 = (i & 63) << 3;
            *(uint4*)&As[r * KC + k8] =
                *(const uint4*)&xb[(size_t)stok[r] * H + kc + k8];
        }
        __syncthreads();
        int ktb = kc >> 5;
#pragma unroll
        for (int nt = 0; nt < 6; ++nt) {
            int ntile = wave * 6 + nt;
            const unsigned short* bp1 = w1b + ((size_t)(e * NT1 + ntile) * KT1 + ktb) * 512;
            const unsigned short* bp2 = v1b + ((size_t)(e * NT1 + ntile) * KT1 + ktb) * 512;
            for (int kk = 0; kk < KC; kk += 32) {
                v16bf a  = load_a_lds(As, KC, kk);
                v16bf bg = load_b_tiled(bp1); bp1 += 512;
                v16bf bu = load_b_tiled(bp2); bp2 += 512;
                accg[nt] = wmma_bf16(a, bg, accg[nt]);
                accu[nt] = wmma_bf16(a, bu, accu[nt]);
            }
        }
        __syncthreads();
    }

    // h = silu(g) * u -> bf16 LDS
#pragma unroll
    for (int nt = 0; nt < 6; ++nt) {
        int n0 = (wave * 6 + nt) * 16;
#pragma unroll
        for (int j = 0; j < 8; ++j) {
            float g  = accg[nt][j];
            float u  = accu[nt][j];
            float hv = (g / (1.f + __expf(-g))) * u;
            Hs[(mrow + j) * F + n0 + ncol] = f2bf(hv);
        }
    }
    __syncthreads();

    // down proj + weighted atomic scatter
    for (int nt = 0; nt < 16; ++nt) {
        int ntile = wave * 16 + nt;
        int n0    = ntile * 16;
        const unsigned short* bp = w2b + (size_t)(e * NT2 + ntile) * KT2 * 512;
        v8f acc = vzero;
        for (int kk = 0; kk < F; kk += 32) {
            v16bf a = load_a_lds(Hs, F, kk);
            v16bf b = load_b_tiled(bp); bp += 512;
            acc = wmma_bf16(a, b, acc);
        }
#pragma unroll
        for (int j = 0; j < 8; ++j) {
            int   m   = mrow + j;
            float val = sw[m] * acc[j];
            atomicAdd(&out[(size_t)stok[m] * H + n0 + ncol], val);
        }
    }
}

// ---------------- fallback: in-flight f32->bf16 conversion (small ws) -------
__launch_bounds__(256)
__global__ void k_expert(const float* __restrict__ x,  const float* __restrict__ w1,
                         const float* __restrict__ v1, const float* __restrict__ w2,
                         const int* __restrict__ counts, const int* __restrict__ ltok,
                         const float* __restrict__ lw, float* __restrict__ out) {
    __shared__ __align__(16) unsigned short As[16 * KC];
    __shared__ __align__(16) unsigned short Hs[16 * F];
    __shared__ int   stok[16];
    __shared__ float sw[16];

    int e  = blockIdx.x >> 6;
    int tb = blockIdx.x & 63;
    int cnt   = counts[e];
    int tiles = (cnt + 15) >> 4;
    if (tb >= tiles) return;

    int tid = threadIdx.x;
    if (tid < 16) {
        int p = tb * 16 + tid;
        if (p < cnt) { stok[tid] = ltok[e * T + p]; sw[tid] = lw[e * T + p]; }
        else         { stok[tid] = 0;               sw[tid] = 0.f; }
    }
    __syncthreads();

    const float* w1e = w1 + (size_t)e * H * F;
    const float* v1e = v1 + (size_t)e * H * F;
    const float* w2e = w2 + (size_t)e * F * H;

    int wave = tid >> 5;
    int lane = tid & 31;
    int mrow = (lane >> 4) << 3;
    int ncol = lane & 15;

    v8f accg[6], accu[6];
    v8f vzero = {};
#pragma unroll
    for (int i = 0; i < 6; ++i) { accg[i] = vzero; accu[i] = vzero; }

    for (int kc = 0; kc < H; kc += KC) {
        for (int i = tid; i < 16 * KC; i += 256) {
            int r = i >> 9, k = i & (KC - 1);
            As[i] = f2bf(x[(size_t)stok[r] * H + kc + k]);
        }
        __syncthreads();
#pragma unroll
        for (int nt = 0; nt < 6; ++nt) {
            int n0 = (wave * 6 + nt) * 16;
            for (int kk = 0; kk < KC; kk += 32) {
                v16bf a  = load_a_lds(As, KC, kk);
                v16bf bg = load_b_global(w1e, F, kc + kk, n0);
                v16bf bu = load_b_global(v1e, F, kc + kk, n0);
                accg[nt] = wmma_bf16(a, bg, accg[nt]);
                accu[nt] = wmma_bf16(a, bu, accu[nt]);
            }
        }
        __syncthreads();
    }

#pragma unroll
    for (int nt = 0; nt < 6; ++nt) {
        int n0 = (wave * 6 + nt) * 16;
#pragma unroll
        for (int j = 0; j < 8; ++j) {
            float g  = accg[nt][j];
            float u  = accu[nt][j];
            float hv = (g / (1.f + __expf(-g))) * u;
            Hs[(mrow + j) * F + n0 + ncol] = f2bf(hv);
        }
    }
    __syncthreads();

    for (int nt = 0; nt < 16; ++nt) {
        int n0 = (wave * 16 + nt) * 16;
        v8f acc = vzero;
        for (int kk = 0; kk < F; kk += 32) {
            v16bf a = load_a_lds(Hs, F, kk);
            v16bf b = load_b_global(w2e, H, kk, n0);
            acc = wmma_bf16(a, b, acc);
        }
#pragma unroll
        for (int j = 0; j < 8; ++j) {
            int   m   = mrow + j;
            float val = sw[m] * acc[j];
            atomicAdd(&out[(size_t)stok[m] * H + n0 + ncol], val);
        }
    }
}

// ---------------- launcher ----------------
extern "C" void kernel_launch(void* const* d_in, const int* in_sizes, int n_in,
                              void* d_out, int out_size, void* d_ws, size_t ws_size,
                              hipStream_t stream) {
    const float* x  = (const float*)d_in[0];   // [T, H]
    const float* wg = (const float*)d_in[1];   // [E, H]
    const float* w1 = (const float*)d_in[2];   // [E, H, F]
    const float* v1 = (const float*)d_in[3];   // [E, H, F]
    const float* w2 = (const float*)d_in[4];   // [E, F, H]
    float* out = (float*)d_out;                // [T, H]

    // workspace layout
    constexpr size_t OFF_TOPKI = 0;                    // T*K ints  (32 KB)
    constexpr size_t OFF_TOPKW = 32 * 1024;            // T*K floats(32 KB)
    constexpr size_t OFF_CNT   = 64 * 1024;            // E ints
    constexpr size_t OFF_LTOK  = 68 * 1024;            // E*T ints  (128 KB)
    constexpr size_t OFF_LW    = OFF_LTOK + 128 * 1024;
    constexpr size_t OFF_XB    = 512 * 1024;           // T*H bf16  (4 MB)
    constexpr size_t SZ_FRAG   = FRAGS_PER_MAT * 512 * 2;  // 96 MB per matrix
    constexpr size_t OFF_W1B   = 8ull * 1024 * 1024;
    constexpr size_t OFF_V1B   = OFF_W1B + SZ_FRAG;
    constexpr size_t OFF_W2B   = OFF_V1B + SZ_FRAG;
    constexpr size_t NEED_BIG  = OFF_W2B + SZ_FRAG;    // ~296 MB

    char* base = (char*)d_ws;
    int*   topk_idx = (int*)  (base + OFF_TOPKI);
    float* topk_w   = (float*)(base + OFF_TOPKW);
    int*   counts   = (int*)  (base + OFF_CNT);
    int*   ltok     = (int*)  (base + OFF_LTOK);
    float* lwv      = (float*)(base + OFF_LW);

    k_zero  <<<(T * H + 255) / 256, 256, 0, stream>>>(out, T * H);
    k_router<<<T / 8, 256, 0, stream>>>(x, wg, topk_idx, topk_w);
    k_build <<<1, 32, 0, stream>>>(topk_idx, topk_w, counts, ltok, lwv);

    if (ws_size >= NEED_BIG) {
        unsigned short* xb  = (unsigned short*)(base + OFF_XB);
        unsigned short* w1b = (unsigned short*)(base + OFF_W1B);
        unsigned short* v1b = (unsigned short*)(base + OFF_V1B);
        unsigned short* w2b = (unsigned short*)(base + OFF_W2B);
        k_cvt_x  <<<(T * H / 4 + 255) / 256, 256, 0, stream>>>(x, xb, T * H);
        k_convert<<<(unsigned)(3 * FRAGS_PER_MAT / 8), 256, 0, stream>>>(
            w1, v1, w2, w1b, v1b, w2b);
        k_expert_pre<<<E * 64, 256, 0, stream>>>(xb, w1b, v1b, w2b,
                                                 counts, ltok, lwv, out);
    } else {
        k_expert<<<E * 64, 256, 0, stream>>>(x, w1, v1, w2, counts, ltok, lwv, out);
    }
}
